// VQ_TC_Model_ERA5_33045478375524
// MI455X (gfx1250) — compile-verified, matrix operand order checked by reading, not run on hardware
//
#include <hip/hip_runtime.h>
#include <hip/hip_bf16.h>

// ---------------------------------------------------------------------------
// Problem constants (from reference)
// ---------------------------------------------------------------------------
#define B_   4096
#define C_   2
#define K_   2048
#define D_   512
#define P_   8
#define H_   4
#define NT_  64
#define HID_ 2048
#define OUT_ 2
#define FPD_ 4096   // P_*D_
#define DH_  128    // D_/H_
#define BETA_ 0.25f

typedef __attribute__((ext_vector_type(16))) _Float16 v16h;
typedef __attribute__((ext_vector_type(8)))  _Float16 v8h;
typedef __attribute__((ext_vector_type(4)))  _Float16 v4h;
typedef __attribute__((ext_vector_type(8)))  float    v8f;

// ---------------------------------------------------------------------------
// WMMA GEMM: C[M,N] = alpha*(A @ B) + colAdd[N] + rowAdd[M] (+ residual[M,N])
//   Block tile 64(M) x 128(N), 256 threads = 8 wave32 waves.
//   Each wave owns a 32x32 tile -> 2x2 accumulators -> 4 v_wmma per K-step.
//   K staged 32/step through double-buffered LDS in f16.
//   B is stored TRANSPOSED in LDS (Bs[n][k]) so both A and B fragments are
//   two ds_load_b128 per fragment (v16h elems 0..7 = K kh..kh+7 contiguous,
//   elems 8..15 = K 16+kh..16+kh+7; ISA 7.12.2 wave32 layout).
// ---------------------------------------------------------------------------
#define TM 64
#define TN 128
#define LDSA_S 40   // halves; 80B row stride -> 16B aligned, b128 conflict-free
#define LDSB_S 40

__device__ inline v16h frag_lds(const _Float16* base16, int stride) {
  const int lane = threadIdx.x & 31;
  const _Float16* row = base16 + (lane & 15) * stride;
  const int kh = (lane >> 4) << 3;           // lanes 16..31: K+8 block
  const v8h lo = *(const v8h*)(row + kh);       // K = kh .. kh+7
  const v8h hi = *(const v8h*)(row + kh + 16);  // K = 16+kh .. 16+kh+7
  return __builtin_shufflevector(lo, hi, 0, 1, 2, 3, 4, 5, 6, 7,
                                         8, 9, 10, 11, 12, 13, 14, 15);
}

__device__ inline v4h cvt4(float4 v, float m) {
  v4h h;
  h[0] = (_Float16)(v.x * m); h[1] = (_Float16)(v.y * m);
  h[2] = (_Float16)(v.z * m); h[3] = (_Float16)(v.w * m);
  return h;
}

template<int TRANSB>
__global__ __launch_bounds__(256)
void gemm_wmma(const float* __restrict__ A, int lda,
               const float* __restrict__ Bm, int ldb,
               float* __restrict__ C, int M, int N, int K, float alpha,
               const float* __restrict__ colAdd,
               const float* __restrict__ rowAdd, int rowAddStride,
               const float* __restrict__ residual)
{
  __shared__ _Float16 As[2][TM * LDSA_S];
  __shared__ _Float16 Bs[2][TN * LDSB_S];

  const int tid = threadIdx.x;
  const int m0  = blockIdx.y * TM;
  const int n0  = blockIdx.x * TN;
  const int wid = tid >> 5;
  const int wy  = wid >> 2;        // 0..1 -> M offset wy*32
  const int wx  = wid & 3;         // 0..3 -> N offset wx*32

  // ---- staging coordinates (straight-line, fully vectorized) ----
  const int arow = tid >> 3;             // 0..31 (and +32)
  const int kq   = (tid & 7) << 2;       // 0,4,...,28

  const float* Arow0 = A + (size_t)(m0 + arow) * lda;
  const float* Arow1 = A + (size_t)(m0 + arow + 32) * lda;

  int   bn[4];
  float bmask[4];
  const float* Bp[4];
#pragma unroll
  for (int j = 0; j < 4; ++j) {
    const int n = arow + 32 * j;               // 0..127
    const int g = n0 + n;
    const int gc = (g < N) ? g : (N - 1);      // clamp: loads stay in-bounds
    bn[j] = n;
    bmask[j] = (g < N) ? 1.f : 0.f;            // zero-fill ragged columns
    Bp[j] = TRANSB ? (Bm + (size_t)gc * ldb) : (Bm + gc);
  }

  float4 ra[2], rb[4];

  auto load_tile = [&](int k0) {
    ra[0] = *(const float4*)(Arow0 + k0 + kq);
    ra[1] = *(const float4*)(Arow1 + k0 + kq);
#pragma unroll
    for (int j = 0; j < 4; ++j) {
      if (TRANSB) {
        rb[j] = *(const float4*)(Bp[j] + k0 + kq);   // contiguous in K
      } else {
        const float* p = Bp[j] + (size_t)(k0 + kq) * ldb;
        rb[j].x = p[0];
        rb[j].y = p[(size_t)ldb];
        rb[j].z = p[(size_t)2 * ldb];
        rb[j].w = p[(size_t)3 * ldb];
      }
    }
  };

  auto store_tile = [&](int buf) {
    *(v4h*)&As[buf][(size_t)arow * LDSA_S + kq]        = cvt4(ra[0], 1.f);
    *(v4h*)&As[buf][(size_t)(arow + 32) * LDSA_S + kq] = cvt4(ra[1], 1.f);
#pragma unroll
    for (int j = 0; j < 4; ++j)     // transposed: Bs[n][k]
      *(v4h*)&Bs[buf][(size_t)bn[j] * LDSB_S + kq] = cvt4(rb[j], bmask[j]);
  };

  v8f acc[2][2];
#pragma unroll
  for (int mi = 0; mi < 2; ++mi)
#pragma unroll
    for (int ni = 0; ni < 2; ++ni)
      acc[mi][ni] = v8f{0.f, 0.f, 0.f, 0.f, 0.f, 0.f, 0.f, 0.f};

  const int nT = K >> 5;
  load_tile(0);
  store_tile(0);
  __syncthreads();

  for (int t = 0; t < nT; ++t) {
    const int cur = t & 1;
    if (t + 1 < nT) load_tile((t + 1) << 5);            // overlap with WMMA
    if (t + 2 < nT) {                                    // global_prefetch_b8
      __builtin_prefetch(Arow0 + ((t + 2) << 5) + kq, 0, 1);
      __builtin_prefetch(Arow1 + ((t + 2) << 5) + kq, 0, 1);
    }

    v16h af[2], bf[2];
    af[0] = frag_lds(&As[cur][(wy * 32 +  0) * LDSA_S], LDSA_S);
    af[1] = frag_lds(&As[cur][(wy * 32 + 16) * LDSA_S], LDSA_S);
    bf[0] = frag_lds(&Bs[cur][(wx * 32 +  0) * LDSB_S], LDSB_S);
    bf[1] = frag_lds(&Bs[cur][(wx * 32 + 16) * LDSB_S], LDSB_S);

#pragma unroll
    for (int mi = 0; mi < 2; ++mi)
#pragma unroll
      for (int ni = 0; ni < 2; ++ni)
        acc[mi][ni] = __builtin_amdgcn_wmma_f32_16x16x32_f16(
            false, af[mi], false, bf[ni], (short)0, acc[mi][ni], false, false);

    if (t + 1 < nT) store_tile(cur ^ 1);   // other buffer: no race with readers
    __syncthreads();
  }

  // ---- epilogue (C/D layout: VGPR r -> rows r / r+8 across lane halves) ----
  const int lane = tid & 31;
  const int cn   = lane & 15;
  const int rb8  = (lane >> 4) << 3;
#pragma unroll
  for (int mi = 0; mi < 2; ++mi) {
#pragma unroll
    for (int ni = 0; ni < 2; ++ni) {
      const int gcol = n0 + wx * 32 + ni * 16 + cn;
      if (gcol < N) {
#pragma unroll
        for (int r = 0; r < 8; ++r) {
          const int grow = m0 + wy * 32 + mi * 16 + rb8 + r;
          float v = alpha * acc[mi][ni][r];
          if (colAdd)   v += colAdd[gcol];
          if (rowAdd)   v += rowAdd[(size_t)grow * rowAddStride];
          if (residual) v += residual[(size_t)grow * N + gcol];
          C[(size_t)grow * N + gcol] = v;
        }
      }
    }
  }
}

// ---------------------------------------------------------------------------
// Row sum-of-squares: out[r] = sum_d X[r*D + d]^2    (block = 128 per row)
// ---------------------------------------------------------------------------
__global__ __launch_bounds__(128)
void row_sumsq(const float* __restrict__ X, float* __restrict__ out, int Dn) {
  __shared__ float red[128];
  const int r = blockIdx.x;
  const float* row = X + (size_t)r * Dn;
  float s = 0.f;
  for (int d = threadIdx.x; d < Dn; d += 128) { const float v = row[d]; s += v * v; }
  red[threadIdx.x] = s; __syncthreads();
  for (int st = 64; st > 0; st >>= 1) {
    if (threadIdx.x < st) red[threadIdx.x] += red[threadIdx.x + st];
    __syncthreads();
  }
  if (threadIdx.x == 0) out[r] = red[0];
}

// ---------------------------------------------------------------------------
// VQ post-processing (per (c,b) row of dist): argmin -> enc,
// target = softmax(1/dist), logp at inds, quantization MSE.
// ---------------------------------------------------------------------------
__device__ inline float block_sum(float v, float* red) {
  const int tid = threadIdx.x;
  red[tid] = v; __syncthreads();
  for (int st = 128; st > 0; st >>= 1) {
    if (tid < st) red[tid] += red[tid + st];
    __syncthreads();
  }
  const float r = red[0]; __syncthreads();
  return r;
}

__global__ __launch_bounds__(256)
void vq_post(const float* __restrict__ dist,       // [C][B][K]
             const float* __restrict__ embed,      // [B][C][D]
             const float* __restrict__ codebooks,  // [C][K][D]
             const int*   __restrict__ inds,       // [B][C]
             float* __restrict__ target,           // [B][C][K] (into d_out)
             float* __restrict__ accv)
{
  __shared__ float redf[256];
  __shared__ int   redi[256];
  const int r = blockIdx.x;           // c*B + b (matches dist layout)
  const int c = r / B_;
  const int b = r % B_;
  const int tid = threadIdx.x;
  const float* drow = dist + (size_t)r * K_;

  float dloc[K_ / 256];
  float dmin = 3.4e38f; int imin = 0; float smax = -3.4e38f;
  int cnt = 0;
  for (int k = tid; k < K_; k += 256) {
    const float d = drow[k];
    dloc[cnt++] = d;
    if (d < dmin) { dmin = d; imin = k; }
    const float s = 1.f / d;
    if (s > smax) smax = s;
  }
  redf[tid] = dmin; redi[tid] = imin; __syncthreads();
  for (int st = 128; st > 0; st >>= 1) {
    if (tid < st && redf[tid + st] < redf[tid]) {
      redf[tid] = redf[tid + st]; redi[tid] = redi[tid + st];
    }
    __syncthreads();
  }
  const int enc = redi[0]; __syncthreads();

  redf[tid] = smax; __syncthreads();
  for (int st = 128; st > 0; st >>= 1) {
    if (tid < st) redf[tid] = fmaxf(redf[tid], redf[tid + st]);
    __syncthreads();
  }
  const float ms = redf[0]; __syncthreads();

  float z1 = 0.f; cnt = 0;
  for (int k = tid; k < K_; k += 256) z1 += __expf(1.f / dloc[cnt++] - ms);
  const float Z1 = block_sum(z1, redf);
  const float invZ1 = 1.f / Z1;

  const int jidx = inds[b * C_ + c];
  float z2 = 0.f, tj = 0.f; cnt = 0;
  for (int k = tid; k < K_; k += 256) {
    const float t = __expf(1.f / dloc[cnt++] - ms) * invZ1;
    target[((size_t)b * C_ + c) * K_ + k] = t;
    z2 += __expf(t);
    if (k == jidx) tj = t;
  }
  const float Z2 = block_sum(z2, redf);
  const float TJ = block_sum(tj, redf);
  if (tid == 0) atomicAdd(&accv[2 + c], TJ - __logf(Z2));

  const float* crow = codebooks + ((size_t)c * K_ + enc) * D_;
  const float* erow = embed     + ((size_t)b * C_ + c) * D_;
  float qs = 0.f;
  for (int d = tid; d < D_; d += 256) {
    const float df = crow[d] - erow[d];
    qs += df * df;
  }
  const float QS = block_sum(qs, redf);
  if (tid == 0) atomicAdd(&accv[c], QS);
}

__global__ void vq_init_acc(float* accv) {
  if (threadIdx.x < 4) accv[threadIdx.x] = 0.f;
}

__global__ void vq_finalize(const float* __restrict__ accv,
                            float* __restrict__ out_scalar) {
  if (threadIdx.x == 0) {
    float v = 0.f;
    for (int c = 0; c < C_; ++c) {
      const float msec = accv[c] / ((float)B_ * (float)D_);
      const float tl   = -accv[2 + c] / (float)B_;
      v += (1.f + BETA_) * msec + tl;   // BETA*commit + embl + tl
    }
    out_scalar[0] = v / (float)B_;
  }
}

// ---------------------------------------------------------------------------
// Attention core per (b,h): 8x64 scores, softmax, @v.  <1% of total FLOPs.
// ---------------------------------------------------------------------------
__global__ __launch_bounds__(64)
void attention(const float* __restrict__ Q, const float* __restrict__ Kb,
               const float* __restrict__ Vb, float* __restrict__ Y)
{
  __shared__ float qs[P_ * DH_];
  __shared__ float sc[P_ * NT_];
  const int bh = blockIdx.x;
  const int b = bh / H_, h = bh % H_;
  const int tid = threadIdx.x;

  for (int i = tid; i < P_ * DH_; i += 64) {
    const int p = i / DH_, d = i % DH_;
    qs[p * DH_ + d] = Q[((size_t)b * P_ + p) * D_ + h * DH_ + d];
  }
  __syncthreads();

  {
    const int n = tid;   // NT_ == 64 == blockDim
    const float* krow = Kb + ((size_t)b * NT_ + n) * D_ + h * DH_;
    float a[P_];
#pragma unroll
    for (int p = 0; p < P_; ++p) a[p] = 0.f;
    for (int d = 0; d < DH_; ++d) {
      const float kv = krow[d];
#pragma unroll
      for (int p = 0; p < P_; ++p) a[p] += qs[p * DH_ + d] * kv;
    }
    const float scale = rsqrtf((float)DH_);
#pragma unroll
    for (int p = 0; p < P_; ++p) sc[p * NT_ + n] = a[p] * scale;
  }
  __syncthreads();

  if (tid < P_) {
    float mx = -3.4e38f;
    for (int n = 0; n < NT_; ++n) mx = fmaxf(mx, sc[tid * NT_ + n]);
    float s = 0.f;
    for (int n = 0; n < NT_; ++n) {
      const float e = __expf(sc[tid * NT_ + n] - mx);
      sc[tid * NT_ + n] = e; s += e;
    }
    const float inv = 1.f / s;
    for (int n = 0; n < NT_; ++n) sc[tid * NT_ + n] *= inv;
  }
  __syncthreads();

  for (int dd = tid; dd < DH_; dd += 64) {
    float a[P_];
#pragma unroll
    for (int p = 0; p < P_; ++p) a[p] = 0.f;
    for (int n = 0; n < NT_; ++n) {
      const float vv = Vb[((size_t)b * NT_ + n) * D_ + h * DH_ + dd];
#pragma unroll
      for (int p = 0; p < P_; ++p) a[p] += sc[p * NT_ + n] * vv;
    }
#pragma unroll
    for (int p = 0; p < P_; ++p)
      Y[((size_t)b * P_ + p) * D_ + h * DH_ + dd] = a[p];
  }
}

// ---------------------------------------------------------------------------
// In-place LayerNorm over rows of length Dn (block = 256 per row)
// ---------------------------------------------------------------------------
__global__ __launch_bounds__(256)
void layernorm_ip(float* __restrict__ X, const float* __restrict__ g,
                  const float* __restrict__ bb, int Dn)
{
  __shared__ float red[256];
  float* row = X + (size_t)blockIdx.x * Dn;
  const int tid = threadIdx.x;
  float s = 0.f, ss = 0.f;
  for (int d = tid; d < Dn; d += 256) { const float v = row[d]; s += v; ss += v * v; }
  const float S  = block_sum(s, red);
  const float SS = block_sum(ss, red);
  const float mu = S / (float)Dn;
  const float var = SS / (float)Dn - mu * mu;
  const float rstd = rsqrtf(var + 1e-5f);
  for (int d = tid; d < Dn; d += 256)
    row[d] = (row[d] - mu) * rstd * g[d] + bb[d];
}

// ---------------------------------------------------------------------------
// Launch
// ---------------------------------------------------------------------------
extern "C" void kernel_launch(void* const* d_in, const int* in_sizes, int n_in,
                              void* d_out, int out_size, void* d_ws, size_t ws_size,
                              hipStream_t stream) {
  const float* embed = (const float*)d_in[0];
  const float* cbook = (const float*)d_in[1];
  const int*   inds  = (const int*)  d_in[2];
  const float* x     = (const float*)d_in[3];
  const float* img   = (const float*)d_in[4];
  const float* Wq = (const float*)d_in[5];  const float* bq = (const float*)d_in[6];
  const float* Wk = (const float*)d_in[7];  const float* bk = (const float*)d_in[8];
  const float* Wv = (const float*)d_in[9];  const float* bv = (const float*)d_in[10];
  const float* Wo = (const float*)d_in[11]; const float* bo = (const float*)d_in[12];
  const float* ln1g = (const float*)d_in[13]; const float* ln1b = (const float*)d_in[14];
  const float* Wl = (const float*)d_in[15]; const float* bl = (const float*)d_in[16];
  const float* ln2g = (const float*)d_in[17]; const float* ln2b = (const float*)d_in[18];
  const float* W1 = (const float*)d_in[19]; const float* b1 = (const float*)d_in[20];
  const float* W2 = (const float*)d_in[21]; const float* b2 = (const float*)d_in[22];
  const float* W3 = (const float*)d_in[23]; const float* b3 = (const float*)d_in[24];

  float* outp   = (float*)d_out;                       // [B,OUT]
  float* target = outp + (size_t)B_ * OUT_;            // [B,C,K]
  float* vqout  = target + (size_t)B_ * C_ * K_;       // scalar

  float* ws = (float*)d_ws;
  size_t off = 0;
  float* dist = ws + off; off += (size_t)C_ * B_ * K_;     // 16.7M
  float* Qb   = ws + off; off += (size_t)B_ * P_ * D_;     // 16.7M
  float* Kbuf = ws + off; off += (size_t)B_ * NT_ * D_;    // 134M
  float* Vbuf = ws + off; off += (size_t)B_ * NT_ * D_;    // 134M
  float* Yb   = ws + off; off += (size_t)B_ * P_ * D_;     // 16.7M
  float* Hb   = ws + off; off += (size_t)B_ * FPD_;        // 16.7M
  float* H2   = ws + off; off += (size_t)B_ * FPD_;        // 16.7M
  float* H3   = ws + off; off += (size_t)B_ * HID_;        // 8.4M
  float* H4   = ws + off; off += (size_t)B_ * HID_;        // 8.4M
  float* e2   = ws + off; off += (size_t)B_ * C_;
  float* w2   = ws + off; off += (size_t)C_ * K_;
  float* accv = ws + off; off += 4;

  const dim3 blk(256);
  auto gemm = [&](const float* A, int lda, const float* Bm, int ldb, int transB,
                  float* Cp, int M, int N, int K, float alpha,
                  const float* colAdd, const float* rowAdd, int rs,
                  const float* resid) {
    dim3 grid((N + TN - 1) / TN, M / TM);
    if (transB)
      hipLaunchKernelGGL((gemm_wmma<1>), grid, blk, 0, stream,
                         A, lda, Bm, ldb, Cp, M, N, K, alpha, colAdd, rowAdd, rs, resid);
    else
      hipLaunchKernelGGL((gemm_wmma<0>), grid, blk, 0, stream,
                         A, lda, Bm, ldb, Cp, M, N, K, alpha, colAdd, rowAdd, rs, resid);
  };

  // ---- VQ ----
  hipLaunchKernelGGL(vq_init_acc, dim3(1), dim3(32), 0, stream, accv);
  hipLaunchKernelGGL(row_sumsq, dim3(B_ * C_), dim3(128), 0, stream, embed, e2, D_);
  hipLaunchKernelGGL(row_sumsq, dim3(C_ * K_), dim3(128), 0, stream, cbook, w2, D_);
  for (int c = 0; c < C_; ++c) {
    // dist[c] = e2[:,c] + w2[c,:] - 2 * embed[:,c,:] @ codebooks[c]^T
    gemm(embed + (size_t)c * D_, C_ * D_,
         cbook + (size_t)c * K_ * D_, D_, /*transB=*/1,
         dist + (size_t)c * B_ * K_, B_, K_, D_, -2.f,
         w2 + (size_t)c * K_, e2 + c, C_, nullptr);
  }
  hipLaunchKernelGGL(vq_post, dim3(C_ * B_), dim3(256), 0, stream,
                     dist, embed, cbook, inds, target, accv);
  hipLaunchKernelGGL(vq_finalize, dim3(1), dim3(32), 0, stream, accv, vqout);

  // ---- attention projections ----
  gemm(x,   D_, Wq, D_, 0, Qb,   B_ * P_,  D_, D_, 1.f, bq, nullptr, 0, nullptr);
  gemm(img, D_, Wk, D_, 0, Kbuf, B_ * NT_, D_, D_, 1.f, bk, nullptr, 0, nullptr);
  gemm(img, D_, Wv, D_, 0, Vbuf, B_ * NT_, D_, D_, 1.f, bv, nullptr, 0, nullptr);
  hipLaunchKernelGGL(attention, dim3(B_ * H_), dim3(64), 0, stream, Qb, Kbuf, Vbuf, Yb);
  gemm(Yb, D_, Wo, D_, 0, Hb, B_ * P_, D_, D_, 1.f, bo, nullptr, 0, nullptr);

  // ---- MLP head (Hb viewed as [B, FPD]) ----
  hipLaunchKernelGGL(layernorm_ip, dim3(B_), dim3(256), 0, stream, Hb, ln1g, ln1b, FPD_);
  gemm(Hb, FPD_, Wl, FPD_, 0, H2, B_, FPD_, FPD_, 1.f, bl, nullptr, 0, nullptr);
  hipLaunchKernelGGL(layernorm_ip, dim3(B_), dim3(256), 0, stream, H2, ln2g, ln2b, FPD_);
  gemm(H2, FPD_, W1, HID_, 0, H3, B_, HID_, FPD_, 1.f, b1, nullptr, 0, nullptr);
  gemm(H3, HID_, W2, HID_, 0, H4, B_, HID_, HID_, 1.f, b2, nullptr, 0, H3);  // residual
  gemm(H4, HID_, W3, OUT_, 0, outp, B_, OUT_, HID_, 1.f, b3, nullptr, 0, nullptr);
}
